// FDoG_44143673868792
// MI455X (gfx1250) — matrix operand order repaired and checked by reading, not compiled, not run on hardware
//
#include <hip/hip_runtime.h>

// ---------------------------------------------------------------------------
// FDoG for MI455X (gfx1250, wave32).
// Working set ~64MB fp32 -> resident in 192MB L2; all stages latency-bound.
// DoG 21-tap weighted reduction is cast onto v_wmma_f32_16x16x32_f16
// (A = replicated weights 16x32, B = 32-tap x 16-pixel sample tile from LDS).
// Horizontal ETF pass stages its 286-float row strip into LDS, using
// global_load_async_to_lds_b32 (ASYNCcnt) when the toolchain exposes it.
// ---------------------------------------------------------------------------

typedef _Float16 v16h __attribute__((ext_vector_type(16)));
typedef float    v8f  __attribute__((ext_vector_type(8)));

#define DEVFN __device__ __forceinline__

constexpr int   B_        = 2;
constexpr int   Y_        = 1024;
constexpr int   X_        = 1024;
constexpr int   YX_       = Y_ * X_;
constexpr int   NPIX      = B_ * YX_;       // 2 Mi pixels
constexpr int   MU        = 15;
constexpr int   ETF_ITERS = 3;
constexpr int   FDOG_ITERS= 3;
constexpr float SIGMA_C   = 2.0f;
constexpr float SIGMA_M   = 5.0f;
constexpr float RHO       = 0.99f;
constexpr float TAU       = 0.7f;
constexpr float SIGMA_S   = SIGMA_C * 1.6f; // 3.2
constexpr int   MAX_T     = 10;             // ceil(3*SIGMA_S)
constexpr int   MAX_S     = 25;             // floor(5*SIGMA_M)

// Optional CDNA5 async global->LDS path (guarded: cannot break the build
// if the builtins are absent on this toolchain).
#if defined(__has_builtin)
#  if __has_builtin(__builtin_amdgcn_global_load_async_to_lds_b32) && \
      __has_builtin(__builtin_amdgcn_s_wait_asynccnt)
#    define HAVE_ASYNC_LDS 1
#  endif
#endif
#ifndef HAVE_ASYNC_LDS
#  define HAVE_ASYNC_LDS 0
#endif

#if HAVE_ASYNC_LDS
// Builtin signature (from hipcc diagnostics): (AS1 int*, AS3 int*, imm, imm)
#define ASYNC_G2L_B32(gsrc, ldst)                                              \
  __builtin_amdgcn_global_load_async_to_lds_b32(                               \
      (__attribute__((address_space(1))) int*)const_cast<float*>(gsrc),        \
      (__attribute__((address_space(3))) int*)(ldst), 0, 0)
#endif

DEVFN float gpdf(float x, float sigma) {
  // exp(-x^2 / (2 s^2)) / (sqrt(2 pi) s)
  return __expf(-x * x / (2.0f * sigma * sigma)) * (0.3989422804014327f / sigma);
}

DEVFN float dog_weight(int k) {  // k in [0,31]; taps t = k - MAX_T, valid k<=20
  if (k > 2 * MAX_T) return 0.0f;
  float t = (float)(k - MAX_T);
  return gpdf(t, SIGMA_C) - RHO * gpdf(t, SIGMA_S);
}

// ---------------------------------------------------------------------------
// Stage 1: gray = mean over 3 channels
// ---------------------------------------------------------------------------
__global__ __launch_bounds__(256) void k_gray(const float* __restrict__ img,
                                              float* __restrict__ gray) {
  int p = blockIdx.x * 256 + threadIdx.x;
  int bb = p / YX_;
  int r  = p - bb * YX_;
  const float* base = img + (size_t)bb * 3 * YX_ + r;
  gray[p] = (base[0] + base[YX_] + base[2 * YX_]) * (1.0f / 3.0f);
}

// ---------------------------------------------------------------------------
// Stage 2: Sobel -> raw magnitude, normalized tangent; block-reduce max(mag)
// ---------------------------------------------------------------------------
__global__ __launch_bounds__(256) void k_sobel(const float* __restrict__ gray,
                                               float* __restrict__ mag,
                                               float* __restrict__ tg0,
                                               float* __restrict__ tg1,
                                               unsigned int* __restrict__ maxbits) {
  int p  = blockIdx.x * 256 + threadIdx.x;
  int bb = p / YX_;
  int r  = p - bb * YX_;
  int y  = r / X_;
  int x  = r - y * X_;
  auto g = [&](int yy, int xx) -> float {
    return (yy < 0 || yy >= Y_ || xx < 0 || xx >= X_) ? 0.0f
                                                      : gray[bb * YX_ + yy * X_ + xx];
  };
  float a = g(y - 1, x - 1), b = g(y - 1, x), c = g(y - 1, x + 1);
  float d = g(y,     x - 1),                  f = g(y,     x + 1);
  float h = g(y + 1, x - 1), i = g(y + 1, x), j = g(y + 1, x + 1);
  float gy = -a - 2.0f * b - c + h + 2.0f * i + j;   // ker k
  float gx = -a - 2.0f * d - h + c + 2.0f * f + j;   // ker k.T
  float m  = sqrtf(gy * gy + gx * gx);
  mag[p] = m;
  float tn = (m == 0.0f) ? 1.0f : m;
  tg0[p] = -gx / tn;   // tang ch0 = -sob_x
  tg1[p] =  gy / tn;   // tang ch1 =  sob_y

  __shared__ float smax[256];
  int tid = threadIdx.x;
  smax[tid] = m;
  __syncthreads();
  for (int s = 128; s > 0; s >>= 1) {
    if (tid < s) smax[tid] = fmaxf(smax[tid], smax[tid + s]);
    __syncthreads();
  }
  if (tid == 0) atomicMax(maxbits, __float_as_uint(smax[0]));  // m >= 0
}

__global__ __launch_bounds__(256) void k_magnorm(float* __restrict__ mag,
                                                 const unsigned int* __restrict__ maxbits) {
  int p = blockIdx.x * 256 + threadIdx.x;
  float mx = __uint_as_float(*maxbits);
  if (mx <= 0.0f) mx = 1.0f;
  mag[p] = mag[p] / mx;
}

// ---------------------------------------------------------------------------
// Stage 3: one ETF smoothing pass. ORI=0 vertical taps, ORI=1 horizontal taps.
// Horizontal pass stages a 286-float strip of (t0,t1,mag) in LDS.
// ---------------------------------------------------------------------------
template <int ORI>
__global__ __launch_bounds__(256) void k_etf(const float* __restrict__ ti0,
                                             const float* __restrict__ ti1,
                                             const float* __restrict__ mag,
                                             float* __restrict__ to0,
                                             float* __restrict__ to1) {
  constexpr int STRIP = 256 + 2 * MU;  // 286
  __shared__ float s0[STRIP], s1[STRIP], sm[STRIP];

  int p, bb, y, x;
  if (ORI == 1) {
    int bx   = blockIdx.x;
    int tile = bx & 3;            // X_/256 == 4
    int row  = (bx >> 2) & (Y_ - 1);
    bb       = bx >> 12;          // / (4 * Y_)
    int x0   = tile << 8;
    x        = x0 + threadIdx.x;
    y        = row;
    p        = bb * YX_ + y * X_ + x;
    int rowb = bb * YX_ + y * X_;
    for (int i = threadIdx.x; i < STRIP; i += 256) {
      int gx = x0 - MU + i;
      if (gx >= 0 && gx < X_) {
#if HAVE_ASYNC_LDS
        ASYNC_G2L_B32(ti0 + rowb + gx, &s0[i]);
        ASYNC_G2L_B32(ti1 + rowb + gx, &s1[i]);
        ASYNC_G2L_B32(mag + rowb + gx, &sm[i]);
#else
        s0[i] = ti0[rowb + gx];
        s1[i] = ti1[rowb + gx];
        sm[i] = mag[rowb + gx];
#endif
      } else {
        s0[i] = 0.0f; s1[i] = 0.0f; sm[i] = 0.0f;  // zero padding
      }
    }
#if HAVE_ASYNC_LDS
    __builtin_amdgcn_s_wait_asynccnt(0);
#endif
    __syncthreads();
  } else {
    p  = blockIdx.x * 256 + threadIdx.x;
    bb = p / YX_;
    int r = p - bb * YX_;
    y = r / X_;
    x = r - y * X_;
  }

  float tx0, tx1, mc;
  if (ORI == 1) {
    tx0 = s0[threadIdx.x + MU];
    tx1 = s1[threadIdx.x + MU];
    mc  = sm[threadIdx.x + MU];
  } else {
    tx0 = ti0[p];
    tx1 = ti1[p];
    mc  = mag[p];
  }

  float a0 = 0.0f, a1 = 0.0f;
  if (ORI == 0) {
    __builtin_prefetch(ti0 + p - MU * X_, 0, 1);  // global_prefetch_b8 on gfx1250
    __builtin_prefetch(ti0 + p + MU * X_, 0, 1);
  }
#pragma unroll 1
  for (int off = -MU; off <= MU; ++off) {
    float ty0, ty1, sy;
    if (ORI == 1) {
      int i = threadIdx.x + MU + off;
      ty0 = s0[i]; ty1 = s1[i]; sy = sm[i];
    } else {
      int yy = y + off;
      if (yy < 0 || yy >= Y_) continue;   // padded-zero taps contribute 0
      int q = bb * YX_ + yy * X_ + x;
      ty0 = ti0[q]; ty1 = ti1[q]; sy = mag[q];
    }
    float w = (tanhf(sy - mc) + 1.0f) * (tx0 * ty0 + tx1 * ty1) * 0.5f;
    a0 += ty0 * w;
    a1 += ty1 * w;
  }
  float n = sqrtf(a0 * a0 + a1 * a1);
  if (n == 0.0f) n = 1.0f;
  to0[p] = a0 / n;
  to1[p] = a1 / n;
}

// ---------------------------------------------------------------------------
// Stage 4a: img = min(img, fdog)
// ---------------------------------------------------------------------------
__global__ __launch_bounds__(256) void k_min(float* __restrict__ img,
                                             const float* __restrict__ fdog) {
  int p = blockIdx.x * 256 + threadIdx.x;
  img[p] = fminf(img[p], fdog[p]);
}

// ---------------------------------------------------------------------------
// Stage 4b: perpendicular DoG profile via WMMA.
// Each wave: 32 pixels, each gathers 21 taps (padded to K=32) -> fp16 tile in
// LDS -> two v_wmma_f32_16x16x32_f16 with A = replicated Gaussian-difference
// weights give dog[] for all 32 lanes.
// ---------------------------------------------------------------------------
__global__ __launch_bounds__(256) void k_dog(const float* __restrict__ img,
                                             const float* __restrict__ tg0,
                                             const float* __restrict__ tg1,
                                             float* __restrict__ dog) {
  __shared__ _Float16 s_samp[8][32][32];  // [wave][pixel-lane][tap], 16 KB

  int tid  = threadIdx.x;
  int wave = tid >> 5;
  int lane = tid & 31;
  int p    = blockIdx.x * 256 + tid;
  int bb   = p / YX_;
  int r    = p - bb * YX_;
  int y    = r / X_;
  int x    = r - y * X_;

  float e0 = tg0[p], e1 = tg1[p];
  float per0 = -e1;   // perp ch0 (rows)   = -etf_x
  float per1 =  e0;   // perp ch1 (cols)   =  etf_y

#pragma unroll 1
  for (int k = 0; k < 32; ++k) {
    float v = 0.0f;
    if (k <= 2 * MAX_T) {
      float t  = (float)(k - MAX_T);
      float py = fminf(fmaxf((float)y + per0 * t, 0.0f), (float)(Y_ - 1));
      float px = fminf(fmaxf((float)x + per1 * t, 0.0f), (float)(X_ - 1));
      int iy = __float2int_rn(py);   // round-half-even, matches jnp.round
      int ix = __float2int_rn(px);
      v = img[bb * YX_ + iy * X_ + ix];
    }
    s_samp[wave][lane][k] = (_Float16)v;
  }
  __syncthreads();

  // A fragment: 16x32 fp16, every row = DoG weight vector (contiguous K:
  // lanes 0-15 hold K {0..15}, lanes 16-31 hold K {16..31}).
  int hi = (lane >> 4) & 1;
  v16h afrag;
#pragma unroll
  for (int j = 0; j < 8; ++j) {
    int kb = 16 * hi + 2 * j;
    afrag[2 * j]     = (_Float16)dog_weight(kb);
    afrag[2 * j + 1] = (_Float16)dog_weight(kb + 1);
  }

  // B fragments: 32x16 fp16, column n = pixel n's 32 taps.
  int n = lane & 15;
  v16h bfrag0, bfrag1;
#pragma unroll
  for (int j = 0; j < 8; ++j) {
    int k = 16 * hi + 2 * j;
    bfrag0[2 * j]     = s_samp[wave][n][k];
    bfrag0[2 * j + 1] = s_samp[wave][n][k + 1];
    bfrag1[2 * j]     = s_samp[wave][16 + n][k];
    bfrag1[2 * j + 1] = s_samp[wave][16 + n][k + 1];
  }

  v8f c0 = {};
  v8f c1 = {};
  c0 = __builtin_amdgcn_wmma_f32_16x16x32_f16(false, afrag, false, bfrag0,
                                              (short)0, c0, false, false);
  c1 = __builtin_amdgcn_wmma_f32_16x16x32_f16(false, afrag, false, bfrag1,
                                              (short)0, c1, false, false);

  // D rows are identical; VGPR0 holds (M=0,N=lane) / (M=8,N=lane-16).
  float val = (lane < 16) ? c0[0] : c1[0];

  float tw = 0.0f;
#pragma unroll 1
  for (int t = -MAX_T; t <= MAX_T; ++t) tw += dog_weight(t + MAX_T);
  dog[p] = val / tw;
}

// ---------------------------------------------------------------------------
// Stage 4c: streamline integral along ETF (sequential gathers) + threshold.
// ---------------------------------------------------------------------------
__global__ __launch_bounds__(256) void k_trace(const float* __restrict__ dog,
                                               const float* __restrict__ tg0,
                                               const float* __restrict__ tg1,
                                               float* __restrict__ fdog,
                                               float* __restrict__ out) {
  int p  = blockIdx.x * 256 + threadIdx.x;
  int bb = p / YX_;
  int r  = p - bb * YX_;
  int y  = r / X_;
  int x  = r - y * X_;

  float w0  = gpdf(0.0f, SIGMA_M);
  float acc = dog[p] * w0;
  float tw  = w0;
#pragma unroll 1
  for (int s = 1; s <= MAX_S; ++s) tw += 2.0f * gpdf((float)s, SIGMA_M);

#pragma unroll 1
  for (int d = 0; d < 2; ++d) {
    float dir = d ? 1.0f : -1.0f;
    float py = (float)y, px = (float)x;
    int iy = y, ix = x;
#pragma unroll 1
    for (int s = 1; s <= MAX_S; ++s) {
      int q = bb * YX_ + iy * X_ + ix;
      float e0 = tg0[q], e1 = tg1[q];
      py = fminf(fmaxf(py + e0 * dir, 0.0f), (float)(Y_ - 1));
      px = fminf(fmaxf(px + e1 * dir, 0.0f), (float)(X_ - 1));
      iy = __float2int_rn(py);
      ix = __float2int_rn(px);
      float f = dog[bb * YX_ + iy * X_ + ix];
      acc += f * gpdf((float)s, SIGMA_M);
    }
  }
  float v   = acc / tw;
  float res = ((v < 0.0f) && (1.0f + tanhf(v) < TAU)) ? 0.0f : 1.0f;
  fdog[p] = res;
  if (out) out[p] = res;
}

// ---------------------------------------------------------------------------
// Host-side orchestration (graph-capture safe: only kernel launches and
// hipMemsetAsync on `stream`).
// ---------------------------------------------------------------------------
extern "C" void kernel_launch(void* const* d_in, const int* in_sizes, int n_in,
                              void* d_out, int out_size, void* d_ws, size_t ws_size,
                              hipStream_t stream) {
  (void)in_sizes; (void)n_in; (void)out_size; (void)ws_size;
  const float* images = (const float*)d_in[0];
  float* ws = (float*)d_ws;

  float* img   = ws + 0 * (size_t)NPIX;  // gray, then min-accumulated
  float* mag   = ws + 1 * (size_t)NPIX;
  float* ta0   = ws + 2 * (size_t)NPIX;  // tangent / ETF ping
  float* ta1   = ws + 3 * (size_t)NPIX;
  float* tb0   = ws + 4 * (size_t)NPIX;  // ETF pong
  float* tb1   = ws + 5 * (size_t)NPIX;
  float* dogb  = ws + 6 * (size_t)NPIX;
  float* fdogb = ws + 7 * (size_t)NPIX;
  unsigned int* maxb = (unsigned int*)(ws + 8 * (size_t)NPIX);
  float* out = (float*)d_out;

  dim3 blk(256);
  dim3 grd(NPIX / 256);  // 8192 blocks, exact

  (void)hipMemsetAsync(maxb, 0, sizeof(unsigned int), stream);
  k_gray   <<<grd, blk, 0, stream>>>(images, img);
  k_sobel  <<<grd, blk, 0, stream>>>(img, mag, ta0, ta1, maxb);
  k_magnorm<<<grd, blk, 0, stream>>>(mag, maxb);

  for (int it = 0; it < ETF_ITERS; ++it) {
    k_etf<0><<<grd, blk, 0, stream>>>(ta0, ta1, mag, tb0, tb1);  // vertical
    k_etf<1><<<grd, blk, 0, stream>>>(tb0, tb1, mag, ta0, ta1);  // horizontal
  }

  for (int it = 0; it < FDOG_ITERS; ++it) {
    if (it) k_min<<<grd, blk, 0, stream>>>(img, fdogb);
    k_dog  <<<grd, blk, 0, stream>>>(img, ta0, ta1, dogb);
    k_trace<<<grd, blk, 0, stream>>>(dogb, ta0, ta1, fdogb,
                                     (it == FDOG_ITERS - 1) ? out : nullptr);
  }
}
